// HypergraphModel_16226386444663
// MI455X (gfx1250) — matrix operand order repaired and checked by reading, not compiled, not run on hardware
//
#include <hip/hip_runtime.h>
#include <hip/hip_bf16.h>

typedef __attribute__((ext_vector_type(16))) _Float16 v16h;
typedef __attribute__((ext_vector_type(8)))  float    v8f;

#define NN   100000   // nodes
#define NE   100000   // hyperedges
#define NP   1600000  // pins
#define HID  64
#define NEG  0.01f

// ---------------------------------------------------------------- utilities
__global__ void zero_f32(float* __restrict__ p, int n) {
    int i = blockIdx.x * blockDim.x + threadIdx.x;
    if (i < n) p[i] = 0.0f;
}

// ---------------------------------------------------------------- degrees
__global__ void degree_kernel(const int* __restrict__ hidx,
                              float* __restrict__ deg_n,
                              float* __restrict__ deg_e) {
    int p = blockIdx.x * blockDim.x + threadIdx.x;
    if (p >= NP) return;
    atomicAdd(&deg_n[hidx[p]], 1.0f);
    atomicAdd(&deg_e[hidx[NP + p]], 1.0f);
}

__global__ void invert_deg(float* __restrict__ Dinv, float* __restrict__ Binv) {
    int i = blockIdx.x * blockDim.x + threadIdx.x;
    if (i >= NN) return;
    float d = Dinv[i]; Dinv[i] = (d > 0.0f) ? 1.0f / d : 0.0f;
    float e = Binv[i]; Binv[i] = (e > 0.0f) ? 1.0f / e : 0.0f;
}

// ---------------------------------------------------------------- WMMA GEMM
// Y[M x 64] = X[M x KDIM] @ W[KDIM x 64], f16 inputs, f32 accumulate.
// One wave -> one 16-row tile, all 64 output columns (4 n-tiles).
// Weights staged transposed ([n][k]) as f16 in LDS so each lane's B operand
// (16 contiguous k) is a single aligned 32B ds_load_b128 pair.
template <int KDIM>
__global__ void gemm_wmma(const float* __restrict__ X,
                          const float* __restrict__ W,
                          float* __restrict__ Y, int ntiles) {
    __shared__ alignas(32) _Float16 wl[64 * KDIM];
    for (int i = threadIdx.x; i < 64 * KDIM; i += blockDim.x) {
        int n = i / KDIM, k = i % KDIM;
        wl[i] = (_Float16)W[k * 64 + n];
    }
    __syncthreads();

    const int wave = threadIdx.x >> 5;
    const int lane = threadIdx.x & 31;
    const int tile = blockIdx.x * (blockDim.x >> 5) + wave;
    if (tile >= ntiles) return;             // wave-uniform exit, EXEC stays all-1s

    const int half = lane >> 4;             // 0: lanes 0-15, 1: lanes 16-31
    const int rlo  = lane & 15;
    const size_t row = (size_t)tile * 16 + rlo;

    v8f acc[4] = {};
#pragma unroll
    for (int ks = 0; ks < KDIM / 32; ++ks) {
        // A operand: row `row`, k = ks*32 + half*16 + {0..15}
        const float* xr = X + row * KDIM + ks * 32 + half * 16;
        v16h a;
#pragma unroll
        for (int e = 0; e < 8; ++e) {
            a[e]     = (_Float16)xr[e];
            a[8 + e] = (_Float16)xr[8 + e];
        }
#pragma unroll
        for (int t = 0; t < 4; ++t) {
            const int ncol = t * 16 + rlo;
            // B operand: column ncol, k = ks*32 + half*16 + e (contiguous)
            const v16h b = *(const v16h*)&wl[ncol * KDIM + ks * 32 + half * 16];
            acc[t] = __builtin_amdgcn_wmma_f32_16x16x32_f16(
                false, a, false, b, (short)0, acc[t], false, false);
        }
    }
#pragma unroll
    for (int t = 0; t < 4; ++t) {
        const int ncol = t * 16 + rlo;
#pragma unroll
        for (int r = 0; r < 8; ++r) {
            const int m = r + half * 8;     // C/D VGPR layout
            Y[((size_t)tile * 16 + m) * 64 + ncol] = acc[t][r];
        }
    }
}

// ---------------------------------------------------------------- pin traffic
// 16 threads per pin; each handles a float4 chunk of the 64-feature row.
// Phase 1 (node->edge) applies the 1/|e| scale at contribution time, which
// removes a full 25.6MB read+write pass per conv versus scaling afterwards.
__global__ void scatter_pins_scaled(const float* __restrict__ src,
                                    float* __restrict__ dst,
                                    const int* __restrict__ gidx,
                                    const int* __restrict__ sidx,
                                    const float* __restrict__ dscale) {
    int t = blockIdx.x * blockDim.x + threadIdx.x;
    int p = t >> 4;
    if (p >= NP) return;
    int c = (t & 15) * 4;
    int sdst = sidx[p];
    float sc = dscale[sdst];                     // Binv[e], L0-resident
    const float4 v = *(const float4*)&src[(size_t)gidx[p] * 64 + c];
    float* d = &dst[(size_t)sdst * 64 + c];
    atomicAdd(d + 0, v.x * sc);
    atomicAdd(d + 1, v.y * sc);
    atomicAdd(d + 2, v.z * sc);
    atomicAdd(d + 3, v.w * sc);
}

// Phase 2 (edge->node), unscaled: Dinv folded into the node epilogues.
__global__ void scatter_pins(const float* __restrict__ src,
                             float* __restrict__ dst,
                             const int* __restrict__ gidx,
                             const int* __restrict__ sidx) {
    int t = blockIdx.x * blockDim.x + threadIdx.x;
    int p = t >> 4;
    if (p >= NP) return;
    int c = (t & 15) * 4;
    const float4 v = *(const float4*)&src[(size_t)gidx[p] * 64 + c];
    float* d = &dst[(size_t)sidx[p] * 64 + c];
    atomicAdd(d + 0, v.x);
    atomicAdd(d + 1, v.y);
    atomicAdd(d + 2, v.z);
    atomicAdd(d + 3, v.w);
}

// ---------------------------------------------------------------- epilogues
// out = leaky(LN(agg*Dinv + b)); one wave32 per node, 2 feats per lane.
__global__ void pw_ln_leaky(const float* __restrict__ agg,
                            const float* __restrict__ Dinv,
                            const float* __restrict__ b,
                            const float* __restrict__ g,
                            const float* __restrict__ be,
                            float* __restrict__ out) {
    int node = (blockIdx.x * blockDim.x + threadIdx.x) >> 5;
    int lane = threadIdx.x & 31;
    if (node >= NN) return;
    float di = Dinv[node];
    size_t base = (size_t)node * 64;
    float v0 = agg[base + lane]      * di + b[lane];
    float v1 = agg[base + 32 + lane] * di + b[32 + lane];
    float s = v0 + v1, sq = v0 * v0 + v1 * v1;
#pragma unroll
    for (int off = 16; off; off >>= 1) {
        s  += __shfl_xor(s,  off, 32);
        sq += __shfl_xor(sq, off, 32);
    }
    float m = s * (1.0f / 64.0f);
    float var = sq * (1.0f / 64.0f) - m * m;
    float r = rsqrtf(var + 1e-5f);
    float y0 = (v0 - m) * r * g[lane]      + be[lane];
    float y1 = (v1 - m) * r * g[32 + lane] + be[32 + lane];
    out[base + lane]      = (y0 >= 0.0f) ? y0 : NEG * y0;
    out[base + 32 + lane] = (y1 >= 0.0f) ? y1 : NEG * y1;
}

// out = agg*Dinv + b, optional leaky.
__global__ void pw_bias_act(const float* __restrict__ agg,
                            const float* __restrict__ Dinv,
                            const float* __restrict__ b,
                            float* __restrict__ out, int total, int leaky) {
    int i = blockIdx.x * blockDim.x + threadIdx.x;
    if (i >= total) return;
    float v = agg[i] * Dinv[i >> 6] + b[i & 63];
    if (leaky) v = (v >= 0.0f) ? v : NEG * v;
    out[i] = v;
}

// ---------------------------------------------------------------- decoder
__global__ void decoder_kernel(const float* __restrict__ mu,
                               const float* __restrict__ dW1, const float* __restrict__ db1,
                               const float* __restrict__ dW2, const float* __restrict__ db2,
                               const float* __restrict__ dW3, const float* __restrict__ db3,
                               float* __restrict__ dec, float* __restrict__ norm) {
    __shared__ float w1[64 * 32], w2[32 * 8], w3[8], bb1[32], bb2[8], bb3;
    for (int i = threadIdx.x; i < 64 * 32; i += blockDim.x) w1[i] = dW1[i];
    for (int i = threadIdx.x; i < 32 * 8;  i += blockDim.x) w2[i] = dW2[i];
    if (threadIdx.x < 8)  w3[threadIdx.x]  = dW3[threadIdx.x];
    if (threadIdx.x < 32) bb1[threadIdx.x] = db1[threadIdx.x];
    if (threadIdx.x < 8)  bb2[threadIdx.x] = db2[threadIdx.x];
    if (threadIdx.x == 0) bb3 = db3[0];
    __syncthreads();

    int node = blockIdx.x * blockDim.x + threadIdx.x;
    float o = 0.0f;
    if (node < NN) {
        float z[64];
        const float* zr = mu + (size_t)node * 64;
#pragma unroll
        for (int k = 0; k < 64; ++k) z[k] = zr[k];
        float h1[32];
        for (int j = 0; j < 32; ++j) {
            float s = bb1[j];
            for (int k = 0; k < 64; ++k) s += z[k] * w1[k * 32 + j];
            h1[j] = (s >= 0.0f) ? s : NEG * s;
        }
        float h2[8];
        for (int j = 0; j < 8; ++j) {
            float s = bb2[j];
            for (int k = 0; k < 32; ++k) s += h1[k] * w2[k * 8 + j];
            h2[j] = (s >= 0.0f) ? s : NEG * s;
        }
        o = bb3;
        for (int k = 0; k < 8; ++k) o += h2[k] * w3[k];
        dec[node] = o;
    }
    float sq = (node < NN) ? o * o : 0.0f;
#pragma unroll
    for (int off = 16; off; off >>= 1) sq += __shfl_xor(sq, off, 32);
    if ((threadIdx.x & 31) == 0) atomicAdd(norm, sq);
}

__global__ void finalize_kernel(const float* __restrict__ dec,
                                const float* __restrict__ norm,
                                float* __restrict__ out) {
    int i = blockIdx.x * blockDim.x + threadIdx.x;
    if (i >= NN) return;
    float nv = sqrtf(*norm);
    float d = dec[i] / fmaxf(nv, 1e-8f);
    out[i]      = d;   // z_orth
    out[NN + i] = d;   // mu_orth (z == mu in eval)
}

// ---------------------------------------------------------------- launcher
extern "C" void kernel_launch(void* const* d_in, const int* in_sizes, int n_in,
                              void* d_out, int out_size, void* d_ws, size_t ws_size,
                              hipStream_t stream) {
    const float* x    = (const float*)d_in[0];
    const int*   hidx = (const int*)d_in[1];        // [0..NP)=nidx, [NP..2NP)=eidx
    const float* W1   = (const float*)d_in[2];
    const float* b1   = (const float*)d_in[3];
    const float* g1   = (const float*)d_in[4];
    const float* be1  = (const float*)d_in[5];
    const float* W2   = (const float*)d_in[6];
    const float* b2   = (const float*)d_in[7];
    const float* Wmu  = (const float*)d_in[8];
    const float* bmu  = (const float*)d_in[9];
    const float* Wlv  = (const float*)d_in[10];
    const float* blv  = (const float*)d_in[11];
    const float* dW1  = (const float*)d_in[12];
    const float* db1  = (const float*)d_in[13];
    const float* dW2  = (const float*)d_in[14];
    const float* db2  = (const float*)d_in[15];
    const float* dW3  = (const float*)d_in[16];
    const float* db3  = (const float*)d_in[17];
    float* out = (float*)d_out;

    float* w    = (float*)d_ws;
    float* Dinv = w;               w += NN;
    float* Binv = w;               w += NE;
    float* bufA = w;               w += (size_t)NN * 64;  // GEMM out / node agg
    float* bufE = w;               w += (size_t)NE * 64;  // edge accumulator
    float* bufB = w;               w += (size_t)NN * 64;  // persistent hidden
    float* bufC = w;               w += (size_t)NN * 64;  // mu
    float* dec  = w;               w += NN;
    float* norm = w;               w += 1;

    const int* nidx = hidx;
    const int* eidx = hidx + NP;

    const int B = 256;
    const int NTILES   = NN / 16;                    // 6250
    const int gGemm    = (NTILES + 7) / 8;           // 8 waves/block
    const int gElem    = (NN * 64 + B - 1) / B;      // 6.4M elems
    const int gPins16  = (NP * 16 + B - 1) / B;
    const int gPins    = (NP + B - 1) / B;
    const int gNode    = (NN + B - 1) / B;
    const int gNodeW   = (NN * 32 + B - 1) / B;      // one wave per node

    // --- degrees -> Dinv / Binv
    zero_f32<<<(NN + NE + B - 1) / B, B, 0, stream>>>(Dinv, NN + NE);
    degree_kernel<<<gPins, B, 0, stream>>>(hidx, Dinv, Binv);
    invert_deg<<<gNode, B, 0, stream>>>(Dinv, Binv);

    // --- hyperedge conv: src[NNxK] @ W -> bufA -> edge agg (pre-scaled by
    //     Binv) -> node agg back into bufA (Dinv applied in epilogue)
    auto hconv = [&](const float* src, const float* Wt, int Kdim) {
        if (Kdim == 32)
            gemm_wmma<32><<<gGemm, B, 0, stream>>>(src, Wt, bufA, NTILES);
        else
            gemm_wmma<64><<<gGemm, B, 0, stream>>>(src, Wt, bufA, NTILES);
        zero_f32<<<gElem, B, 0, stream>>>(bufE, NE * 64);
        scatter_pins_scaled<<<gPins16, B, 0, stream>>>(bufA, bufE, nidx, eidx,
                                                       Binv);
        zero_f32<<<gElem, B, 0, stream>>>(bufA, NN * 64);
        scatter_pins<<<gPins16, B, 0, stream>>>(bufE, bufA, eidx, nidx);
    };

    // layer 1: conv + LN + leaky
    hconv(x, W1, 32);
    pw_ln_leaky<<<gNodeW, B, 0, stream>>>(bufA, Dinv, b1, g1, be1, bufB);

    // layer 2: conv + leaky
    hconv(bufB, W2, 64);
    pw_bias_act<<<gElem, B, 0, stream>>>(bufA, Dinv, b2, bufB, NN * 64, 1);

    // mu head
    hconv(bufB, Wmu, 64);
    pw_bias_act<<<gElem, B, 0, stream>>>(bufA, Dinv, bmu, bufC, NN * 64, 0);

    // logvar head -> straight into d_out[2*NN ..)
    hconv(bufB, Wlv, 64);
    pw_bias_act<<<gElem, B, 0, stream>>>(bufA, Dinv, blv, out + 2 * NN, NN * 64, 0);

    // decoder + column L2 normalize
    zero_f32<<<1, B, 0, stream>>>(norm, 1);
    decoder_kernel<<<gNode, B, 0, stream>>>(bufC, dW1, db1, dW2, db2, dW3, db3,
                                            dec, norm);
    finalize_kernel<<<gNode, B, 0, stream>>>(dec, norm, out);
}
